// CEEI_29532195127475
// MI455X (gfx1250) — matrix operand order repaired and checked by reading, not compiled, not run on hardware
//
#include <hip/hip_runtime.h>
#include <hip/hip_bf16.h>

typedef __attribute__((ext_vector_type(16))) _Float16 v16h;
typedef __attribute__((ext_vector_type(8)))  _Float16 v8h;
typedef __attribute__((ext_vector_type(8)))  float    v8f;
typedef __attribute__((ext_vector_type(2)))  float    v2f;
typedef __attribute__((ext_vector_type(4)))  unsigned v4u;
typedef __attribute__((ext_vector_type(8)))  unsigned v8u;

#define LAMV   0.1f
#define ALPHAV 0.02f
#define RESV   2.0f

__device__ __forceinline__ v16h cat8(v8h lo, v8h hi) {
    union { v16h v; v8h h[2]; } u;
    u.h[0] = lo; u.h[1] = hi;
    return u.v;
}

// ---------------------------------------------------------------------------
// Prep kernel: write W1/W2/W3 transposed (Wt[n][k] = W[k][n]) as f16 into ws.
// Layout in wt buffer: [0,8192) = W1t (128x64), [8192,24576) = W2t (128x128),
// [24576,40960) = W3t (128x128).  One element per thread.
// ---------------------------------------------------------------------------
__global__ __launch_bounds__(256) void ceei_prep_kernel(
    const float* __restrict__ W1, const float* __restrict__ W2,
    const float* __restrict__ W3, _Float16* __restrict__ wt)
{
    int i = blockIdx.x * 256 + threadIdx.x;
    if (i < 8192) {
        int n = i >> 6, k = i & 63;
        wt[i] = (_Float16)W1[k * 128 + n];
    } else if (i < 8192 + 16384) {
        int j = i - 8192;
        int n = j >> 7, k = j & 127;
        wt[i] = (_Float16)W2[k * 128 + n];
    } else if (i < 8192 + 32768) {
        int j = i - 24576;
        int n = j >> 7, k = j & 127;
        wt[i] = (_Float16)W3[k * 128 + n];
    }
}

// ---------------------------------------------------------------------------
// Kernel 1: MLP (transposed: Hᵀ = Wᵀ·Xᵀ; lane = batch row; 16 rows per wave).
// All three weight blocks are DMA'd into separate LDS buffers via TDM at
// kernel start; TENSORcnt is drained progressively (2 -> 1 -> 0) so W2/W3
// transfers overlap layer-1/2 compute.  Dynamic LDS (~84 KB).
// ---------------------------------------------------------------------------
__global__ __launch_bounds__(256) void ceei_mlp_kernel(
    const float* __restrict__ dv,
    const _Float16* __restrict__ wt,      // pre-transposed f16 weights in ws
    const float* __restrict__ b1, const float* __restrict__ b2,
    const float* __restrict__ b3,
    const float* __restrict__ W4, const float* __restrict__ b4,
    float* __restrict__ cout)
{
    extern __shared__ char smem[];
    _Float16* WtA = (_Float16*)smem;               // 16KB: W1t 128x64
    _Float16* WtB = (_Float16*)(smem + 16384);     // 32KB: W2t 128x128
    _Float16* WtC = (_Float16*)(smem + 49152);     // 32KB: W3t 128x128
    float* B1s = (float*)(smem + 81920);           // 128
    float* B2s = B1s + 128;
    float* B3s = B2s + 128;
    float* W4s = B3s + 128;                        // 256
    float* B4s = W4s + 256;                        // 2

    const int tid  = threadIdx.x;
    const int lane = tid & 31;
    const int n    = lane & 15;   // batch row within tile
    const int g    = lane >> 4;   // half-wave group
    const int wave = tid >> 5;
    const int rowBase = blockIdx.x * 128 + wave * 16;

    // ---- TDM: DMA `elems` f16 values from global `src` into LDS at `dst` ----
    auto tdmLoad = [&](const _Float16* src, unsigned elems, _Float16* dst) {
        unsigned ldsa = (unsigned)(uintptr_t)dst;
        unsigned long long ga = (unsigned long long)(uintptr_t)src;
        v4u g0;
        g0[0] = 1u;                               // count=1, user desc
        g0[1] = ldsa;                             // lds_addr [63:32]
        g0[2] = (unsigned)ga;                     // global_addr [95:64]
        g0[3] = ((unsigned)(ga >> 32) & 0x01FFFFFFu) | (2u << 30); // addr hi | type=2
        v8u g1;
        g1[0] = 0x00010000u;                      // wg_mask=0, data_size=1 (2B)
        g1[1] = (elems & 0xffffu) << 16;          // tensor_dim0[15:0] @63:48
        g1[2] = (elems >> 16) | (1u << 16);       // dim0[31:16], tensor_dim1=1
        g1[3] = (elems & 0xffffu) << 16;          // tile_dim0 @127:112
        g1[4] = 1u;                               // tile_dim1=1, tile_dim2=0
        g1[5] = elems;                            // tensor_dim0_stride low
        g1[6] = 0u;
        g1[7] = 0u;
        asm volatile("tensor_load_to_lds %0, %1" :: "s"(g0), "s"(g1) : "memory");
    };

    // Issue all three weight DMAs back-to-back (in-order per wave).
    if (wave == 0) {
        tdmLoad(wt,          8192,  WtA);
        tdmLoad(wt + 8192,   16384, WtB);
        tdmLoad(wt + 24576,  16384, WtC);
    }

    // Stage biases / W4 with regular stores (tiny).
    if (tid < 128) {
        B1s[tid] = b1[tid];
        B2s[tid] = b2[tid];
        B3s[tid] = b3[tid];
    }
    W4s[tid] = W4[tid];                   // 256 floats
    if (tid < 2) B4s[tid] = b4[tid];

    v8f  acc[8];
    v16h bfrag[4];

    // ---- B-fragments for layer 1 straight from dv (Xᵀ is the B matrix) ----
    #pragma unroll
    for (int kt = 0; kt < 2; ++kt) {
        const float4* p =
            (const float4*)(dv + (size_t)(rowBase + n) * 64 + kt * 32 + g * 16);
        float4 q0 = p[0], q1 = p[1], q2 = p[2], q3 = p[3];
        float f[16] = { q0.x,q0.y,q0.z,q0.w, q1.x,q1.y,q1.z,q1.w,
                        q2.x,q2.y,q2.z,q2.w, q3.x,q3.y,q3.z,q3.w };
        union { v16h v; _Float16 e[16]; } bu;
        #pragma unroll
        for (int j = 0; j < 16; ++j) bu.e[j] = (_Float16)f[j];
        bfrag[kt] = bu.v;
    }

    auto runLayer = [&](const _Float16* Wt, const float* Bs, int K, int KT,
                        bool makeB) {
        #pragma unroll
        for (int mt = 0; mt < 8; ++mt)
            acc[mt] = (v8f){0.f,0.f,0.f,0.f,0.f,0.f,0.f,0.f};
        for (int kt = 0; kt < KT; ++kt) {
            #pragma unroll
            for (int mt = 0; mt < 8; ++mt) {
                int row = mt * 16 + n;                // A row = output unit
                v8h lo = *(const v8h*)&Wt[row * K + kt * 32 + g * 16];
                v8h hi = *(const v8h*)&Wt[row * K + kt * 32 + g * 16 + 8];
                v16h afrag = cat8(lo, hi);
                acc[mt] = __builtin_amdgcn_wmma_f32_16x16x32_f16(
                    false, afrag, false, bfrag[kt], (short)0, acc[mt],
                    false, false);
            }
        }
        // bias + tanh  (C layout: lane n, VGPR r -> unit mt*16 + g*8 + r)
        #pragma unroll
        for (int mt = 0; mt < 8; ++mt) {
            const float4* bp = (const float4*)&Bs[mt * 16 + g * 8];
            float4 ba = bp[0], bb = bp[1];
            float t[8] = { ba.x,ba.y,ba.z,ba.w, bb.x,bb.y,bb.z,bb.w };
            #pragma unroll
            for (int r = 0; r < 8; ++r)
                acc[mt][r] = tanhf(acc[mt][r] + t[r]);
        }
        if (!makeB) return;
        // C layout -> next layer's B-fragments via half-wave exchange
        #pragma unroll
        for (int kt2 = 0; kt2 < 4; ++kt2) {
            v8f town  = g ? acc[2 * kt2 + 1] : acc[2 * kt2];
            v8f tsend = g ? acc[2 * kt2]     : acc[2 * kt2 + 1];
            float rcvv[8];
            #pragma unroll
            for (int r = 0; r < 8; ++r)
                rcvv[r] = __shfl_xor(tsend[r], 16, 32);
            union { v16h v; _Float16 e[16]; } bu;
            #pragma unroll
            for (int r = 0; r < 8; ++r) {
                float f0 = g ? rcvv[r] : town[r];
                float f1 = g ? town[r] : rcvv[r];
                bu.e[r]     = (_Float16)f0;
                bu.e[8 + r] = (_Float16)f1;
            }
            bfrag[kt2] = bu.v;
        }
    };

    // Drain TENSORcnt progressively; buffers are distinct so no barriers
    // are needed after each layer's compute.
    if (wave == 0) __builtin_amdgcn_s_wait_tensorcnt(2);
    __syncthreads();                      // biases visible + WtA ready
    runLayer(WtA, B1s, 64, 2, true);

    if (wave == 0) __builtin_amdgcn_s_wait_tensorcnt(1);
    __syncthreads();                      // WtB ready
    runLayer(WtB, B2s, 128, 4, true);

    if (wave == 0) __builtin_amdgcn_s_wait_tensorcnt(0);
    __syncthreads();                      // WtC ready
    runLayer(WtC, B3s, 128, 4, false);

    // ---- layer 4: 128 -> 2, plain FMA (each lane holds 64 of 128 units) ----
    float c0 = 0.f, c1 = 0.f;
    #pragma unroll
    for (int mt = 0; mt < 8; ++mt) {
        #pragma unroll
        for (int r = 0; r < 8; ++r) {
            int unit = mt * 16 + g * 8 + r;
            float2 w = *(const float2*)&W4s[unit * 2];
            c0 += acc[mt][r] * w.x;
            c1 += acc[mt][r] * w.y;
        }
    }
    c0 += __shfl_xor(c0, 16, 32);
    c1 += __shfl_xor(c1, 16, 32);
    if (g == 0) {
        float2 cc = make_float2(c0 + B4s[0], c1 + B4s[1]);
        *(float2*)&cout[(size_t)(rowBase + n) * 2] = cc;
    }
}

// ---------------------------------------------------------------------------
// Kernel 2: interior-point prox-gradient solver + final scaling.
// One lane per batch element; D / slacks held as float2 pairs so the
// FMA-dominant loops can lower to V_PK_FMA_F32 (VOP3P packed f32).
// ---------------------------------------------------------------------------
__global__ __launch_bounds__(256) void ceei_solve_kernel(
    const float* __restrict__ dv, const float* __restrict__ cw,
    float* __restrict__ out, int B)
{
    int b = blockIdx.x * 256 + threadIdx.x;
    if (b >= B) return;

    v2f D0p[16], D1p[16], sp[16];
    const float4* p = (const float4*)(dv + (size_t)b * 64);
    #pragma unroll
    for (int q = 0; q < 8; ++q) {
        float4 f = p[q];
        D0p[q*2]   = (v2f){f.x, f.y};
        D0p[q*2+1] = (v2f){f.z, f.w};
    }
    #pragma unroll
    for (int q = 0; q < 8; ++q) {
        float4 f = p[8 + q];
        D1p[q*2]   = (v2f){f.x, f.y};
        D1p[q*2+1] = (v2f){f.z, f.w};
    }

    float c0 = cw[(size_t)b * 2], c1 = cw[(size_t)b * 2 + 1];

    // x0 = 0.5 * min_m( RES / (D0+D1) )
    float mn = 1e30f;
    #pragma unroll
    for (int m = 0; m < 16; ++m) {
        v2f cs = D0p[m] + D1p[m];
        mn = fminf(mn, RESV * __builtin_amdgcn_rcpf(cs.x));
        mn = fminf(mn, RESV * __builtin_amdgcn_rcpf(cs.y));
    }
    float x0 = 0.5f * mn, x1 = x0;

    const float tsv[5] = {1.f, 10.f, 100.f, 1000.f, 10000.f};
    #pragma unroll 1
    for (int stg = 0; stg < 5; ++stg) {
        float invt = __builtin_amdgcn_rcpf(tsv[stg]);
        #pragma unroll 1
        for (int it = 0; it < 100; ++it) {
            v2f x0v = {x0, x0}, x1v = {x1, x1};
            v2f rv  = {RESV, RESV};
            v2f ga0 = {0.f, 0.f}, ga1 = {0.f, 0.f};
            #pragma unroll
            for (int m = 0; m < 16; ++m) {
                v2f sm = rv - x0v * D0p[m] - x1v * D1p[m];
                sp[m] = sm;
                v2f inv = { __builtin_amdgcn_rcpf(sm.x),
                            __builtin_amdgcn_rcpf(sm.y) };
                ga0 += D0p[m] * inv;
                ga1 += D1p[m] * inv;
            }
            float g0 = -__builtin_amdgcn_rcpf(x0) + invt * (ga0.x + ga0.y);
            float g1 = -__builtin_amdgcn_rcpf(x1) + invt * (ga1.x + ga1.y);
            float z0 = x0 - ALPHAV * g0 - c0;
            float z1 = x1 - ALPHAV * g1 - c1;
            float st0 = fmaxf(fabsf(z0) - ALPHAV * LAMV, 0.f);
            float st1 = fmaxf(fabsf(z1) - ALPHAV * LAMV, 0.f);
            float xp0 = c0 + copysignf(st0, z0);
            float xp1 = c1 + copysignf(st1, z1);
            float d0 = xp0 - x0, d1 = xp1 - x1;

            v2f d0v = {d0, d0}, d1v = {d1, d1};
            float thA = 1e30f, thB = 1e30f;        // two independent chains
            #pragma unroll
            for (int m = 0; m < 16; ++m) {
                v2f dd = d0v * D0p[m] + d1v * D1p[m];
                float rA = sp[m].x * __builtin_amdgcn_rcpf(dd.x);
                float rB = sp[m].y * __builtin_amdgcn_rcpf(dd.y);
                thA = (dd.x > 1e-12f) ? fminf(thA, rA) : thA;
                thB = (dd.y > 1e-12f) ? fminf(thB, rB) : thB;
            }
            float th = fminf(thA, thB);
            if (d0 < -1e-12f) th = fminf(th, -x0 * __builtin_amdgcn_rcpf(d0));
            if (d1 < -1e-12f) th = fminf(th, -x1 * __builtin_amdgcn_rcpf(d1));
            float theta = fminf(1.0f, 0.99f * th);
            x0 += theta * d0;
            x1 += theta * d1;
        }
    }

    float4* op = (float4*)(out + (size_t)b * 64);
    #pragma unroll
    for (int q = 0; q < 8; ++q) {
        v2f a = D0p[q*2], bb = D0p[q*2+1];
        op[q] = make_float4(a.x * x0, a.y * x0, bb.x * x0, bb.y * x0);
    }
    #pragma unroll
    for (int q = 0; q < 8; ++q) {
        v2f a = D1p[q*2], bb = D1p[q*2+1];
        op[8 + q] = make_float4(a.x * x1, a.y * x1, bb.x * x1, bb.y * x1);
    }
}

extern "C" void kernel_launch(void* const* d_in, const int* in_sizes, int n_in,
                              void* d_out, int out_size, void* d_ws, size_t ws_size,
                              hipStream_t stream) {
    const float* dv = (const float*)d_in[0];
    const float* W1 = (const float*)d_in[1];
    const float* b1 = (const float*)d_in[2];
    const float* W2 = (const float*)d_in[3];
    const float* b2 = (const float*)d_in[4];
    const float* W3 = (const float*)d_in[5];
    const float* b3 = (const float*)d_in[6];
    const float* W4 = (const float*)d_in[7];
    const float* b4 = (const float*)d_in[8];
    float* out = (float*)d_out;

    int B = in_sizes[0] / 64;                      // 65536

    float*     cw = (float*)d_ws;                          // c[B,2] (512 KB)
    _Float16*  wt = (_Float16*)((char*)d_ws + (size_t)B * 2 * sizeof(float));
                                                           // 40960 f16 (80 KB)

    const size_t mlpLds = 81920 + (128 * 3 + 256 + 2) * sizeof(float); // ~84.5KB

    ceei_prep_kernel<<<160, 256, 0, stream>>>(W1, W2, W3, wt);
    ceei_mlp_kernel<<<B / 128, 256, mlpLds, stream>>>(dv, wt, b1, b2, b3,
                                                      W4, b4, cw);
    ceei_solve_kernel<<<(B + 255) / 256, 256, 0, stream>>>(dv, cw, out, B);
}
